// time_mamba_85194971283781
// MI455X (gfx1250) — matrix operand order repaired and checked by reading, not compiled, not run on hardware
//
#include <hip/hip_runtime.h>

#define D_MODEL 512
#define D_INNER 1024
#define D_STATE 16
#define D_CONV  4
#define DT_RANK 32
#define BATCH   2
#define SEQ     2048
#define ROWS    (BATCH * SEQ)   /* 4096 */

typedef _Float16 half8 __attribute__((ext_vector_type(8)));
typedef _Float16 v16h  __attribute__((ext_vector_type(16)));
typedef float    v8f   __attribute__((ext_vector_type(8)));

// ---------------------------------------------------------------- utilities

__global__ void cvt_f16_kernel(const float* __restrict__ src,
                               _Float16* __restrict__ dst, int n) {
  int i = blockIdx.x * blockDim.x + threadIdx.x;
  if (i < n) dst[i] = (_Float16)src[i];
}

// LayerNorm: one wave32 per 512-wide row; emit f16 for the WMMA GEMM.
__global__ void ln_kernel(const float* __restrict__ x, const float* __restrict__ w,
                          const float* __restrict__ b, _Float16* __restrict__ h) {
  int wave = (blockIdx.x * blockDim.x + threadIdx.x) >> 5;
  int lane = threadIdx.x & 31;
  if (wave >= ROWS) return;
  const float* xr = x + (size_t)wave * D_MODEL;
  float v[16], s = 0.f, q = 0.f;
#pragma unroll
  for (int i = 0; i < 16; ++i) {
    float t = xr[lane + 32 * i];
    v[i] = t; s += t; q += t * t;
  }
#pragma unroll
  for (int m = 16; m >= 1; m >>= 1) {
    s += __shfl_xor(s, m, 32);
    q += __shfl_xor(q, m, 32);
  }
  float mu  = s * (1.f / D_MODEL);
  float var = q * (1.f / D_MODEL) - mu * mu;
  float inv = rsqrtf(var + 1e-5f);
  _Float16* hr = h + (size_t)wave * D_MODEL;
#pragma unroll
  for (int i = 0; i < 16; ++i) {
    int c = lane + 32 * i;
    hr[c] = (_Float16)((v[i] - mu) * inv * w[c] + b[c]);
  }
}

// ----------------------------------------------------------- async LDS copy
// CDNA5 GLOBAL_LOAD_ASYNC_TO_LDS_B128: per enabled lane, copies 16 bytes from
// global memory to LDS. VDST VGPR = LDS byte offset, VADDR = 64-bit address.
// Tracked with ASYNCcnt (per issuing wave); paired with s_wait_asynccnt.
__device__ __forceinline__ void async_copy_b128(unsigned lds_off, const void* gptr) {
  asm volatile("global_load_async_to_lds_b128 %0, %1, off"
               :: "v"(lds_off), "v"((unsigned long long)gptr)
               : "memory");
}

__device__ __forceinline__ unsigned lds_offset_of(const void* p) {
  // Generic pointer to LDS = {shared_aperture, 32-bit wave-relative offset};
  // low 32 bits are exactly the LDS_ADDR the async op consumes (ISA 10.2).
  return (unsigned)(unsigned long long)p;
}

// A fragment: 16x32 f16, lane r = row M, g = lane>>4 selects K-subgroups.
__device__ __forceinline__ v16h load_a_frag(const _Float16* __restrict__ A,
                                            int K, int m0, int r, int g, int k0) {
  const _Float16* pa = A + (size_t)(m0 + r) * K + k0 + 8 * g;
  half8 lo = *(const half8*)pa;        // K = k0+8g .. +7
  half8 hi = *(const half8*)(pa + 16); // K = k0+16+8g .. +7
  v16h af;
#pragma unroll
  for (int i = 0; i < 8; ++i) { af[i] = lo[i]; af[i + 8] = hi[i]; }
  return af;
}

// ------------------------------------------------------------- WMMA GEMM
// C(MxN, f32, row-major) = A(MxK, f16, row-major) * W(NxK, f16, row-major)^T
// Block = 8 waves = 128x64 C tile. The 64-row B tile for each 32-wide K-step
// is staged in LDS (double-buffered) with async global->LDS copies shared by
// all 8 waves; each wave computes one 16x64 strip (4 WMMA n-tiles).
// Steady-state loop is branch-free; last two K-steps are peeled so the
// s_wait_asynccnt counts stay exact. ACT: 0 = none, 1 = softplus(x + bias)
#define B_ROW_H 40  /* halves per LDS row: 32 data + 8 pad = 80B, bank-safe */

template <int ACT>
__global__ void wmma_gemm_kernel(const _Float16* __restrict__ A,
                                 const _Float16* __restrict__ W,
                                 const float* __restrict__ bias,
                                 float* __restrict__ C,
                                 int M, int N, int K) {
  __shared__ _Float16 bsh[2][64 * B_ROW_H];   // 2 x 5120B

  const int ngroups = N >> 6;
  const int mb = blockIdx.x / ngroups;        // 128-row block index
  const int ng = blockIdx.x - mb * ngroups;
  const int n0 = ng << 6;
  const int waveId = threadIdx.x >> 5;
  const int m0 = (mb << 7) + (waveId << 4);
  const int lane = threadIdx.x & 31;
  const int g = lane >> 4;                    // K-half (A,B) / m-half (C)
  const int r = lane & 15;                    // row for A, col for B/C

  // Cooperative B-tile copy: 256 threads x 16B = 64 rows x 64B (32 halves).
  const int crow  = threadIdx.x >> 2;         // 0..63 : weight row within tile
  const int cpart = threadIdx.x & 3;          // 0..3  : 8-half chunk within row
  const _Float16* gsrc = W + (size_t)(n0 + crow) * K + cpart * 8;
  const unsigned loff0 = lds_offset_of(&bsh[0][crow * B_ROW_H + cpart * 8]);
  const unsigned loff1 = lds_offset_of(&bsh[1][crow * B_ROW_H + cpart * 8]);
  // Per-lane LDS read base (half-index) for the B fragments.
  const int rdbase = r * B_ROW_H + 8 * g;

  v8f acc[4] = {};

  // WMMA compute phase for one 32-wide K-step out of LDS buffer `bp`.
  auto compute = [&](const _Float16* bp, v16h af) {
#pragma unroll
    for (int t = 0; t < 4; ++t) {
      const _Float16* pb = bp + rdbase + t * (16 * B_ROW_H);
      half8 b0 = *(const half8*)pb;
      half8 b1 = *(const half8*)(pb + 16);
      v16h bf;
#pragma unroll
      for (int i = 0; i < 8; ++i) { bf[i] = b0[i]; bf[i + 8] = b1[i]; }
      acc[t] = __builtin_amdgcn_wmma_f32_16x16x32_f16(
          /*neg_a=*/false, af, /*neg_b=*/false, bf,
          /*c_mod=*/(short)0, acc[t], /*reuse_a=*/false, /*reuse_b=*/false);
    }
  };

  // Prologue: fill both buffers (copy i lands in buffer i&1).
  async_copy_b128(loff0, gsrc);
  if (K > 32) async_copy_b128(loff1, gsrc + 32);
  v16h a_cur = load_a_frag(A, K, m0, r, g, 0);

  int p = 0;
  // Steady state: iterations 0 .. niter-3 (branch-free body).
  int k0 = 0;
  for (; k0 + 64 < K; k0 += 32) {
    v16h a_nxt = load_a_frag(A, K, m0, r, g, k0 + 32); // overlap with wait
    asm volatile("s_wait_asynccnt 0x1" ::: "memory");  // copy k0 has landed
    __syncthreads();                                   // visible to all waves
    compute(p ? bsh[1] : bsh[0], a_cur);
    __syncthreads();                                   // done reading buffer p
    async_copy_b128(p ? loff1 : loff0, gsrc + k0 + 64);
    a_cur = a_nxt;
    p ^= 1;
  }
  // Tail A (iteration niter-2, only if K >= 64): next A, no new copy issue.
  if (k0 + 32 < K) {
    v16h a_nxt = load_a_frag(A, K, m0, r, g, k0 + 32);
    asm volatile("s_wait_asynccnt 0x1" ::: "memory");
    __syncthreads();
    compute(p ? bsh[1] : bsh[0], a_cur);
    a_cur = a_nxt;
    p ^= 1;
  }
  // Tail B (iteration niter-1): wait for everything, final compute.
  asm volatile("s_wait_asynccnt 0x0" ::: "memory");
  __syncthreads();
  compute(p ? bsh[1] : bsh[0], a_cur);

  // Epilogue. C/D layout: VGPR v -> row m0+v+8g, column n0+16t+r.
#pragma unroll
  for (int t = 0; t < 4; ++t) {
    const int col = n0 + 16 * t + r;
    const float bv = bias ? bias[col] : 0.f;
#pragma unroll
    for (int v = 0; v < 8; ++v) {
      const int row = m0 + v + 8 * g;
      float val = acc[t][v] + bv;
      if (ACT == 1) val = (val > 20.f) ? val : log1pf(__expf(val));
      C[(size_t)row * N + col] = val;
    }
  }
}

// ----------------------------------------------- depthwise conv (4 taps) + SiLU
__global__ void conv_silu_kernel(const float* __restrict__ xz,
                                 const float* __restrict__ cw,
                                 const float* __restrict__ cb,
                                 float* __restrict__ u32,
                                 _Float16* __restrict__ u16) {
  int idx = blockIdx.x * blockDim.x + threadIdx.x;
  if (idx >= ROWS * D_INNER) return;
  int d   = idx & (D_INNER - 1);
  int row = idx >> 10;
  int l = row & (SEQ - 1);
  int b = row >> 11;
  float acc = cb[d];
#pragma unroll
  for (int k = 0; k < D_CONV; ++k) {
    int ls = l + k - (D_CONV - 1);
    if (ls >= 0)
      acc += xz[((size_t)(b * SEQ + ls)) * (2 * D_INNER) + d] * cw[d * D_CONV + k];
  }
  float uv = acc / (1.f + __expf(-acc));   // SiLU
  u32[idx] = uv;
  u16[idx] = (_Float16)uv;
}

// Pack dt-rank slice of dbc (4096x64 f32) into f16 A matrix (4096x32).
__global__ void pack_dtin_kernel(const float* __restrict__ dbc,
                                 _Float16* __restrict__ dtin) {
  int idx = blockIdx.x * blockDim.x + threadIdx.x;
  if (idx >= ROWS * DT_RANK) return;
  int r = idx >> 5, c = idx & 31;
  dtin[idx] = (_Float16)dbc[r * (DT_RANK + 2 * D_STATE) + c];
}

// ------------------------------------------------------------ selective scan
// One 16-lane half-wave per (batch, channel); lane = state index.
__global__ void scan_kernel(const float* __restrict__ dt,
                            const float* __restrict__ dbc,
                            const float* __restrict__ u,
                            const float* __restrict__ xz,
                            const float* __restrict__ A_log,
                            const float* __restrict__ Dp,
                            _Float16* __restrict__ y16) {
  int tid  = blockIdx.x * blockDim.x + threadIdx.x;
  int chan = tid >> 4;                     // 0 .. BATCH*D_INNER-1
  int s    = tid & 15;                     // state index
  if (chan >= BATCH * D_INNER) return;
  int b = chan >> 10;
  int d = chan & (D_INNER - 1);
  const float Aa = -__expf(A_log[d * D_STATE + s]);
  const float Dv = Dp[d];
  float h = 0.f;
  for (int l = 0; l < SEQ; ++l) {
    int row = b * SEQ + l;
    float dtv = dt[(size_t)row * D_INNER + d];
    float uv  = u[(size_t)row * D_INNER + d];
    const float* dr = dbc + (size_t)row * (DT_RANK + 2 * D_STATE);
    float Bv = dr[DT_RANK + s];
    float Cv = dr[DT_RANK + D_STATE + s];
    h = __expf(dtv * Aa) * h + dtv * Bv * uv;
    float p = h * Cv;
    p += __shfl_xor(p, 8, 16);
    p += __shfl_xor(p, 4, 16);
    p += __shfl_xor(p, 2, 16);
    p += __shfl_xor(p, 1, 16);
    if (s == 0) {
      float y  = p + Dv * uv;
      float zv = xz[(size_t)row * (2 * D_INNER) + D_INNER + d];
      y *= zv / (1.f + __expf(-zv));       // * SiLU(z)
      y16[(size_t)row * D_INNER + d] = (_Float16)y;
    }
  }
}

// ------------------------------------------------------------------ launcher

extern "C" void kernel_launch(void* const* d_in, const int* in_sizes, int n_in,
                              void* d_out, int out_size, void* d_ws, size_t ws_size,
                              hipStream_t stream) {
  const float* x      = (const float*)d_in[0];
  const float* ln_w   = (const float*)d_in[1];
  const float* ln_b   = (const float*)d_in[2];
  const float* W_in   = (const float*)d_in[3];
  const float* conv_w = (const float*)d_in[4];
  const float* conv_b = (const float*)d_in[5];
  const float* W_x    = (const float*)d_in[6];
  const float* W_dt   = (const float*)d_in[7];
  const float* b_dt   = (const float*)d_in[8];
  const float* A_log  = (const float*)d_in[9];
  const float* Dp     = (const float*)d_in[10];
  const float* W_out  = (const float*)d_in[11];
  float* out = (float*)d_out;

  char* ws = (char*)d_ws;
  size_t off = 0;
  auto alloc = [&](size_t bytes) -> char* {
    char* p = ws + off;
    off += (bytes + 255) & ~(size_t)255;
    return p;
  };
  _Float16* h16    = (_Float16*)alloc((size_t)ROWS * D_MODEL * 2);
  _Float16* Win16  = (_Float16*)alloc((size_t)2 * D_INNER * D_MODEL * 2);
  _Float16* Wx16   = (_Float16*)alloc((size_t)(DT_RANK + 2 * D_STATE) * D_INNER * 2);
  _Float16* Wdt16  = (_Float16*)alloc((size_t)D_INNER * DT_RANK * 2);
  _Float16* Wout16 = (_Float16*)alloc((size_t)D_MODEL * D_INNER * 2);
  float*    xz     = (float*)   alloc((size_t)ROWS * 2 * D_INNER * 4);
  float*    u32    = (float*)   alloc((size_t)ROWS * D_INNER * 4);
  _Float16* u16    = (_Float16*)alloc((size_t)ROWS * D_INNER * 2);
  float*    dbc    = (float*)   alloc((size_t)ROWS * (DT_RANK + 2 * D_STATE) * 4);
  _Float16* dtin16 = (_Float16*)alloc((size_t)ROWS * DT_RANK * 2);
  float*    dt     = (float*)   alloc((size_t)ROWS * D_INNER * 4);
  _Float16* y16    = (_Float16*)alloc((size_t)ROWS * D_INNER * 2);

  const int TB = 256;
  auto gup = [](int n, int b) { return (n + b - 1) / b; };

  // Weight f32 -> f16 conversions
  cvt_f16_kernel<<<gup(2 * D_INNER * D_MODEL, TB), TB, 0, stream>>>(W_in, Win16, 2 * D_INNER * D_MODEL);
  cvt_f16_kernel<<<gup((DT_RANK + 2 * D_STATE) * D_INNER, TB), TB, 0, stream>>>(W_x, Wx16, (DT_RANK + 2 * D_STATE) * D_INNER);
  cvt_f16_kernel<<<gup(D_INNER * DT_RANK, TB), TB, 0, stream>>>(W_dt, Wdt16, D_INNER * DT_RANK);
  cvt_f16_kernel<<<gup(D_MODEL * D_INNER, TB), TB, 0, stream>>>(W_out, Wout16, D_MODEL * D_INNER);

  // LayerNorm -> h16 (4096 waves)
  ln_kernel<<<gup(ROWS * 32, TB), TB, 0, stream>>>(x, ln_w, ln_b, h16);

  // GEMM1: xz(4096x2048) = h16 * W_in^T (K=512): 32 m-blocks x 32 n-groups
  wmma_gemm_kernel<0><<<(ROWS / 128) * (2 * D_INNER / 64), TB, 0, stream>>>(
      h16, Win16, nullptr, xz, ROWS, 2 * D_INNER, D_MODEL);

  // Depthwise conv + SiLU -> u
  conv_silu_kernel<<<gup(ROWS * D_INNER, TB), TB, 0, stream>>>(xz, conv_w, conv_b, u32, u16);

  // GEMM2: dbc(4096x64) = u16 * W_x^T (K=1024): 32 blocks
  wmma_gemm_kernel<0><<<(ROWS / 128) * ((DT_RANK + 2 * D_STATE) / 64), TB, 0, stream>>>(
      u16, Wx16, nullptr, dbc, ROWS, DT_RANK + 2 * D_STATE, D_INNER);

  // Pack dt-rank slice -> f16
  pack_dtin_kernel<<<gup(ROWS * DT_RANK, TB), TB, 0, stream>>>(dbc, dtin16);

  // GEMM3: dt(4096x1024) = softplus(dtin16 * W_dt^T + b_dt) (K=32): 512 blocks
  wmma_gemm_kernel<1><<<(ROWS / 128) * (D_INNER / 64), TB, 0, stream>>>(
      dtin16, Wdt16, b_dt, dt, ROWS, D_INNER, DT_RANK);

  // Selective scan + D*u + SiLU(z) gate -> y16 (32768 threads)
  scan_kernel<<<gup(BATCH * D_INNER * 16, TB), TB, 0, stream>>>(
      dt, dbc, u32, xz, A_log, Dp, y16);

  // GEMM4: out(4096x512) = y16 * W_out^T (K=1024): 256 blocks
  wmma_gemm_kernel<0><<<(ROWS / 128) * (D_MODEL / 64), TB, 0, stream>>>(
      y16, Wout16, nullptr, out, ROWS, D_MODEL, D_INNER);
}
